// InterpretableMultiHeadAttention_4277787427575
// MI455X (gfx1250) — matrix operand order repaired and checked by reading, not compiled, not run on hardware
//
#include <hip/hip_runtime.h>
#include <hip/hip_bf16.h>

#define N_HEADC 16
#define D_HEADC 64
#define HIDDENC 1024
#define TSEQ    2048
#define BATCHC  4
#define QKV_OUTC ((2 * N_HEADC + 1) * D_HEADC) /* 2112 */
#define BTROWS  (BATCHC * TSEQ)                /* 8192 */

typedef __attribute__((ext_vector_type(16))) __bf16 v16bf;
typedef __attribute__((ext_vector_type(8)))  float  v8f;

union BFVec {
  v16bf v;
  unsigned short u[16];
  uint4 q[2];
};

__device__ __forceinline__ unsigned short f2bf(float f) {
  unsigned int u = __float_as_uint(f);
  u += 0x7FFFu + ((u >> 16) & 1u);   // round to nearest even
  return (unsigned short)(u >> 16);
}

// Load 16 bf16 elements as two 16-byte chunks (p0 -> elements 0..7, p1 -> 8..15).
__device__ __forceinline__ v16bf load16_pair(const unsigned short* p0,
                                             const unsigned short* p1) {
  BFVec t;
  t.q[0] = *reinterpret_cast<const uint4*>(p0);
  t.q[1] = *reinterpret_cast<const uint4*>(p1);
  return t.v;
}

__device__ __forceinline__ v8f wmma_bf16(v16bf a, v16bf b, v8f c) {
  // D = A(16x32) * B(32x16) + C, fp32 accumulate
  return __builtin_amdgcn_wmma_f32_16x16x32_bf16(
      /*neg_a=*/false, a, /*neg_b=*/false, b,
      /*c_mod=*/(short)0, c, /*reuse_a=*/false, /*reuse_b=*/false);
}

// ---------------------------------------------------------------------------
// fp32 -> bf16 conversion (grid-stride)
// ---------------------------------------------------------------------------
__global__ void cvt_f32_bf16(const float* __restrict__ src,
                             unsigned short* __restrict__ dst, int n) {
  int i = blockIdx.x * blockDim.x + threadIdx.x;
  int stride = gridDim.x * blockDim.x;
  for (; i < n; i += stride) dst[i] = f2bf(src[i]);
}

// ---------------------------------------------------------------------------
// QKV GEMM: qkv[m, o] = sum_k x[m,k] * w_qkv[o,k] + b_qkv[o]
// Register-blocked: one wave computes a 32(M) x 64(N) tile = 8 WMMA accums.
// Result split into Q/K ([b,h,t,d] bf16) and shared V ([b,t,d] bf16).
// Each 64-wide N block maps to exactly one head (or V), so the split is
// uniform per wave.
// ---------------------------------------------------------------------------
__global__ __launch_bounds__(256) void qkv_gemm_split(
    const unsigned short* __restrict__ xb, const unsigned short* __restrict__ wb,
    const float* __restrict__ bqkv,
    unsigned short* __restrict__ qb, unsigned short* __restrict__ kb,
    unsigned short* __restrict__ vb) {
  const int NB = QKV_OUTC / 64;  // 33 n-blocks of 4 tiles
  int wave = (blockIdx.x * blockDim.x + threadIdx.x) >> 5;
  int lane = threadIdx.x & 31;
  int mblk = wave / NB, nblk = wave % NB;
  if (mblk >= BTROWS / 32) return;

  int nloc = lane & 15, hi = lane >> 4;
  const unsigned short* arow0 = xb + (size_t)(mblk * 32 + nloc) * HIDDENC;
  const unsigned short* arow1 = arow0 + (size_t)16 * HIDDENC;
  const unsigned short* bcol[4];
#pragma unroll
  for (int t = 0; t < 4; ++t)
    bcol[t] = wb + (size_t)(nblk * 64 + t * 16 + nloc) * HIDDENC;

  v8f acc[2][4] = {};
  for (int k0 = 0; k0 < HIDDENC; k0 += 32) {
    __builtin_prefetch(arow0 + k0 + 256, 0, 3);
    __builtin_prefetch(arow1 + k0 + 256, 0, 3);
    v16bf a0 = load16_pair(arow0 + k0 + 8 * hi, arow0 + k0 + 16 + 8 * hi);
    v16bf a1 = load16_pair(arow1 + k0 + 8 * hi, arow1 + k0 + 16 + 8 * hi);
#pragma unroll
    for (int t = 0; t < 4; ++t) {
      v16bf bm = load16_pair(bcol[t] + k0 + 8 * hi, bcol[t] + k0 + 16 + 8 * hi);
      acc[0][t] = wmma_bf16(a0, bm, acc[0][t]);
      acc[1][t] = wmma_bf16(a1, bm, acc[1][t]);
    }
  }

  // One head (or V) per wave
  unsigned short* dstbase;
  int hh;
  if (nblk < N_HEADC)          { dstbase = qb; hh = nblk; }
  else if (nblk < 2 * N_HEADC) { dstbase = kb; hh = nblk - N_HEADC; }
  else                         { dstbase = vb; hh = 0; }

#pragma unroll
  for (int mi = 0; mi < 2; ++mi) {
#pragma unroll
    for (int t = 0; t < 4; ++t) {
      int d = t * 16 + nloc;               // 0..63 within head
      float bias = bqkv[nblk * 64 + d];
#pragma unroll
      for (int e = 0; e < 8; ++e) {
        int m = mblk * 32 + mi * 16 + 8 * hi + e;   // row in [0, BT)
        int bb = m / TSEQ, tt = m % TSEQ;
        unsigned short bv = f2bf(acc[mi][t][e] + bias);
        if (nblk < 2 * N_HEADC)
          dstbase[(((size_t)bb * N_HEADC + hh) * TSEQ + tt) * D_HEADC + d] = bv;
        else
          dstbase[((size_t)bb * TSEQ + tt) * D_HEADC + d] = bv;
      }
    }
  }
}

// ---------------------------------------------------------------------------
// Flash-style causal attention. One wave (32 threads) per (b, h, 16-row block).
// j-blocks of 32. V panel staged to LDS via async global->LDS copies.
// Writes attn_vec [b,h,t,d] fp32 (second tuple output).
// ---------------------------------------------------------------------------
__global__ __launch_bounds__(32) void flash_attn(
    const unsigned short* __restrict__ qb, const unsigned short* __restrict__ kb,
    const unsigned short* __restrict__ vb, float* __restrict__ attn_vec) {
  __shared__ alignas(16) float sS[16 * 32];
  __shared__ alignas(16) unsigned short sP[16 * 32];
  __shared__ alignas(16) unsigned short sV[32 * 64];   // V panel, row-major
  __shared__ float row_m[16], row_l[16], row_sc[16];

  const int IB = TSEQ / 16;  // 128
  int lane = threadIdx.x;
  int ib = blockIdx.x % IB;
  int h = (blockIdx.x / IB) % N_HEADC;
  int b = blockIdx.x / (IB * N_HEADC);
  int i0 = ib * 16;
  int nloc = lane & 15, hi = lane >> 4;

  const unsigned short* qbase = qb + ((size_t)(b * N_HEADC + h) * TSEQ) * D_HEADC;
  const unsigned short* kbase = kb + ((size_t)(b * N_HEADC + h) * TSEQ) * D_HEADC;
  const unsigned short* vbase = vb + ((size_t)b * TSEQ) * D_HEADC;
  unsigned int ldsV = (unsigned int)(uintptr_t)&sV[0];

  // Q tile (16 rows x 64), A-layout, loaded once
  const unsigned short* qrow = qbase + (size_t)(i0 + nloc) * D_HEADC;
  v16bf aq0 = load16_pair(qrow + 8 * hi, qrow + 16 + 8 * hi);
  v16bf aq1 = load16_pair(qrow + 32 + 8 * hi, qrow + 48 + 8 * hi);

  v8f oacc[4] = {};
  if (lane < 16) {
    row_m[lane] = -1e30f;
    row_l[lane] = 0.f;
  }
  __syncthreads();

  for (int j0 = 0; j0 <= i0 + 15; j0 += 32) {
    // --- async stage V rows [j0, j0+32) x 64 cols (4KB) into LDS ---
    {
      const size_t glim = (size_t)TSEQ * (D_HEADC * 2) - 16;
      int ofs = lane * 16;
#pragma unroll
      for (int t = 0; t < 8; ++t) {
        size_t gofs = (size_t)j0 * (D_HEADC * 2) + (size_t)t * 512 + ofs;
        if (gofs > glim) gofs = glim;            // OOB rows: P == 0 there
        const char* gp = (const char*)vbase + gofs;
        unsigned int lp = ldsV + t * 512 + ofs;
        asm volatile("global_load_async_to_lds_b128 %0, %1, off"
                     :: "v"(lp), "v"(gp) : "memory");
      }
    }

    // --- scores: S(16 x 32) = scale * Q K^T, causal mask ---
#pragma unroll
    for (int s = 0; s < 2; ++s) {
      int jj = j0 + 16 * s + nloc;
      int jcl = (jj < TSEQ) ? jj : (TSEQ - 1);  // clamp OOB K reads (masked anyway)
      const unsigned short* kcol = kbase + (size_t)jcl * D_HEADC;
      v8f sc = {};
      v16bf b0 = load16_pair(kcol + 8 * hi, kcol + 16 + 8 * hi);
      sc = wmma_bf16(aq0, b0, sc);
      v16bf b1 = load16_pair(kcol + 32 + 8 * hi, kcol + 48 + 8 * hi);
      sc = wmma_bf16(aq1, b1, sc);
#pragma unroll
      for (int e = 0; e < 8; ++e) {
        int m = 8 * hi + e;
        float v = sc[e] * 0.125f;            // D_HEAD^-0.5
        if (jj > i0 + m) v = -1e30f;         // causal (also covers jj >= T)
        sS[m * 32 + 16 * s + nloc] = v;
      }
    }
    __syncthreads();

    // --- online softmax (one row per lane, lanes 0..15) ---
    if (lane < 16) {
      int r = lane;
      float tmax = -1e30f;
      for (int n = 0; n < 32; ++n) tmax = fmaxf(tmax, sS[r * 32 + n]);
      float mo = row_m[r];
      float mn = fmaxf(mo, tmax);            // j0 <= i0 always => mn finite
      float scl = __expf(mo - mn);           // 0 on first block, else rescale
      float psum = 0.f;
      for (int n = 0; n < 32; ++n) {
        float p = __expf(sS[r * 32 + n] - mn);  // masked -> exp(-huge) = 0
        psum += p;
        sP[r * 32 + n] = f2bf(p);
      }
      row_l[r] = row_l[r] * scl + psum;
      row_m[r] = mn;
      row_sc[r] = scl;
    }
    __syncthreads();

    // --- rescale accumulators ---
    float f[8];
#pragma unroll
    for (int e = 0; e < 8; ++e) f[e] = row_sc[8 * hi + e];
#pragma unroll
    for (int e = 0; e < 8; ++e) {
      oacc[0][e] *= f[e]; oacc[1][e] *= f[e];
      oacc[2][e] *= f[e]; oacc[3][e] *= f[e];
    }

    // --- wait for V panel, then PV: O(16x64) += P(16x32) * V(32x64) ---
    asm volatile("s_wait_asynccnt 0x0" ::: "memory");
    const unsigned short* pr = sP + (size_t)nloc * 32;
    v16bf pa = load16_pair(pr + 16 * hi, pr + 16 * hi + 8);  // ds_load_b128
#pragma unroll
    for (int cdx = 0; cdx < 4; ++cdx) {
      BFVec bv;
#pragma unroll
      for (int e = 0; e < 16; ++e)           // transpose gather from LDS
        bv.u[e] = sV[(16 * hi + e) * D_HEADC + 16 * cdx + nloc];
      oacc[cdx] = wmma_bf16(pa, bv.v, oacc[cdx]);
    }
    __syncthreads();
  }

  // --- normalize and write attn_vec [b,h,i,d] ---
  float linv[8];
#pragma unroll
  for (int e = 0; e < 8; ++e) linv[e] = 1.f / row_l[8 * hi + e];
#pragma unroll
  for (int cdx = 0; cdx < 4; ++cdx) {
#pragma unroll
    for (int e = 0; e < 8; ++e) {
      int m = 8 * hi + e;
      attn_vec[(((size_t)b * N_HEADC + h) * TSEQ + i0 + m) * D_HEADC +
               16 * cdx + nloc] = oacc[cdx][e] * linv[e];
    }
  }
}

// ---------------------------------------------------------------------------
// Mean over heads -> bf16 [b*t, d]
// ---------------------------------------------------------------------------
__global__ void mean_heads(const float* __restrict__ av,
                           unsigned short* __restrict__ mbuf) {
  int idx = blockIdx.x * blockDim.x + threadIdx.x;
  if (idx >= BATCHC * TSEQ * D_HEADC) return;
  int d = idx % D_HEADC;
  int t = (idx / D_HEADC) % TSEQ;
  int b = idx / (D_HEADC * TSEQ);
  float s = 0.f;
#pragma unroll
  for (int h = 0; h < N_HEADC; ++h)
    s += av[(((size_t)b * N_HEADC + h) * TSEQ + t) * D_HEADC + d];
  mbuf[idx] = f2bf(s * (1.0f / N_HEADC));
}

// ---------------------------------------------------------------------------
// Output projection: out[m, n] = sum_k mavg[m,k] * w_out[n,k] + b_out[n]
// Register-blocked 32(M) x 64(N) per wave, K = 64.
// ---------------------------------------------------------------------------
__global__ __launch_bounds__(256) void out_proj(
    const unsigned short* __restrict__ mbuf, const unsigned short* __restrict__ wob,
    const float* __restrict__ bout, float* __restrict__ out) {
  const int NB = HIDDENC / 64;  // 16
  int wave = (blockIdx.x * blockDim.x + threadIdx.x) >> 5;
  int lane = threadIdx.x & 31;
  int mblk = wave / NB, nblk = wave % NB;
  if (mblk >= BTROWS / 32) return;

  int nloc = lane & 15, hi = lane >> 4;
  const unsigned short* arow0 = mbuf + (size_t)(mblk * 32 + nloc) * D_HEADC;
  const unsigned short* arow1 = arow0 + (size_t)16 * D_HEADC;
  const unsigned short* bcol[4];
#pragma unroll
  for (int t = 0; t < 4; ++t)
    bcol[t] = wob + (size_t)(nblk * 64 + t * 16 + nloc) * D_HEADC;

  v8f acc[2][4] = {};
#pragma unroll
  for (int k0 = 0; k0 < D_HEADC; k0 += 32) {
    v16bf a0 = load16_pair(arow0 + k0 + 8 * hi, arow0 + k0 + 16 + 8 * hi);
    v16bf a1 = load16_pair(arow1 + k0 + 8 * hi, arow1 + k0 + 16 + 8 * hi);
#pragma unroll
    for (int t = 0; t < 4; ++t) {
      v16bf bm = load16_pair(bcol[t] + k0 + 8 * hi, bcol[t] + k0 + 16 + 8 * hi);
      acc[0][t] = wmma_bf16(a0, bm, acc[0][t]);
      acc[1][t] = wmma_bf16(a1, bm, acc[1][t]);
    }
  }

#pragma unroll
  for (int mi = 0; mi < 2; ++mi) {
#pragma unroll
    for (int t = 0; t < 4; ++t) {
      int n = nblk * 64 + t * 16 + nloc;
      float bias = bout[n];
#pragma unroll
      for (int e = 0; e < 8; ++e) {
        int m = mblk * 32 + mi * 16 + 8 * hi + e;
        out[(size_t)m * HIDDENC + n] = acc[mi][t][e] + bias;
      }
    }
  }
}

// ---------------------------------------------------------------------------
extern "C" void kernel_launch(void* const* d_in, const int* in_sizes, int n_in,
                              void* d_out, int out_size, void* d_ws, size_t ws_size,
                              hipStream_t stream) {
  const float* x = (const float*)d_in[0];       // [B, T, HIDDEN]
  const float* w_qkv = (const float*)d_in[1];   // [2112, HIDDEN]
  const float* b_qkv = (const float*)d_in[2];   // [2112]
  const float* w_out = (const float*)d_in[3];   // [HIDDEN, 64]
  const float* b_out = (const float*)d_in[4];   // [HIDDEN]

  float* out_main = (float*)d_out;                       // [B*T, HIDDEN]
  float* attn_vec = out_main + (size_t)BTROWS * HIDDENC; // [B, H, T, D]

  char* ws = (char*)d_ws;
  size_t off = 0;
  auto carve = [&](size_t bytes) -> char* {
    char* p = ws + off;
    off += (bytes + 255) & ~(size_t)255;
    return p;
  };
  unsigned short* xb   = (unsigned short*)carve((size_t)BTROWS * HIDDENC * 2);
  unsigned short* wb   = (unsigned short*)carve((size_t)QKV_OUTC * HIDDENC * 2);
  unsigned short* wob  = (unsigned short*)carve((size_t)HIDDENC * D_HEADC * 2);
  unsigned short* qb   = (unsigned short*)carve((size_t)BATCHC * N_HEADC * TSEQ * D_HEADC * 2);
  unsigned short* kb   = (unsigned short*)carve((size_t)BATCHC * N_HEADC * TSEQ * D_HEADC * 2);
  unsigned short* vb   = (unsigned short*)carve((size_t)BATCHC * TSEQ * D_HEADC * 2);
  unsigned short* mbuf = (unsigned short*)carve((size_t)BTROWS * D_HEADC * 2);

  cvt_f32_bf16<<<2048, 256, 0, stream>>>(x, xb, BTROWS * HIDDENC);
  cvt_f32_bf16<<<2048, 256, 0, stream>>>(w_qkv, wb, QKV_OUTC * HIDDENC);
  cvt_f32_bf16<<<256, 256, 0, stream>>>(w_out, wob, HIDDENC * D_HEADC);

  {
    int waves = (BTROWS / 32) * (QKV_OUTC / 64);  // 256 * 33 = 8448
    qkv_gemm_split<<<waves / 8, 256, 0, stream>>>(xb, wb, b_qkv, qb, kb, vb);
  }
  flash_attn<<<BATCHC * N_HEADC * (TSEQ / 16), 32, 0, stream>>>(qb, kb, vb, attn_vec);
  mean_heads<<<(BTROWS * D_HEADC + 255) / 256, 256, 0, stream>>>(attn_vec, mbuf);
  {
    int waves = (BTROWS / 32) * (HIDDENC / 64);   // 256 * 16 = 4096
    out_proj<<<waves / 8, 256, 0, stream>>>(mbuf, wob, b_out, out_main);
  }
}
